// GroupedQueryAttn_39900246180625
// MI455X (gfx1250) — compile-verified, matrix operand order checked by reading, not run on hardware
//
#include <hip/hip_runtime.h>
#include <hip/hip_bf16.h>

typedef __attribute__((ext_vector_type(16))) __bf16 v16bf;
typedef __attribute__((ext_vector_type(8)))  __bf16 v8bf;
typedef __attribute__((ext_vector_type(8)))  float  v8f;
typedef __attribute__((ext_vector_type(4)))  unsigned int u32x4;
typedef __attribute__((ext_vector_type(8)))  int i32x8;
typedef __attribute__((ext_vector_type(4)))  int i32x4;

#define S_LEN   2048
#define DMODEL  4096
#define NQH     32
#define NKVH    8
#define HDIM    128
#define QKV_N   6144   // DMODEL + 2*KV_DIM
#define QKV_LD  6144

#if __has_builtin(__builtin_amdgcn_tensor_load_to_lds) && __has_builtin(__builtin_amdgcn_s_wait_tensorcnt)
#define HAVE_TDM 1
#else
#define HAVE_TDM 0
#endif

// ---------- bf16 helpers ----------
__device__ __forceinline__ __bf16 tobf(float f) {
  unsigned u = __builtin_bit_cast(unsigned, f);
  u += 0x7FFFu + ((u >> 16) & 1u);
  unsigned short h = (unsigned short)(u >> 16);
  return __builtin_bit_cast(__bf16, h);
}
__device__ __forceinline__ float b2f(__bf16 b) {
  unsigned short h = __builtin_bit_cast(unsigned short, b);
  unsigned u = ((unsigned)h) << 16;
  return __builtin_bit_cast(float, u);
}
__device__ __forceinline__ v8f wmma_bf16(v16bf a, v16bf b, v8f c) {
  return __builtin_amdgcn_wmma_f32_16x16x32_bf16(false, a, false, b, (short)0, c, false, false);
}
__device__ __forceinline__ v8f vzero8() {
  v8f z;
#pragma unroll
  for (int i = 0; i < 8; ++i) z[i] = 0.0f;
  return z;
}

// ---------- async global->LDS 16B copy (CDNA5 ASYNCcnt path) ----------
__device__ __forceinline__ void cp_async16(void* lds, const void* g) {
  unsigned l = (unsigned)(size_t)lds;                 // LDS byte offset (low 32 bits)
  unsigned long long ga = (unsigned long long)(size_t)g;
  asm volatile("global_load_async_to_lds_b128 %0, %1, off"
               :: "v"(l), "v"(ga) : "memory");
}
__device__ __forceinline__ void wait_async() {
  asm volatile("s_wait_asynccnt 0" ::: "memory");
}

#if HAVE_TDM
// ---------- TDM: 2-D tile (tile0 x tile1 elements, 2B each) global -> LDS ----------
// g1w0 carries data_size/pad fields; tensor dims/strides in elements.
__device__ __forceinline__ void tdm_load_2d(const void* gaddr, void* lds,
                                            unsigned g1w0, unsigned tdim0, unsigned tdim1,
                                            unsigned tile0, unsigned tile1, unsigned stride0) {
  unsigned long long ga = (unsigned long long)(size_t)gaddr;
  u32x4 g0;
  g0[0] = 1u;                                          // count=1, user descriptor
  g0[1] = (unsigned)(size_t)lds;                       // lds_addr (bytes)
  g0[2] = (unsigned)ga;                                // global_addr[31:0]
  g0[3] = (unsigned)((ga >> 32) & 0x1FFFFFFull) | 0x80000000u;  // addr[56:32] | type=2
  i32x8 g1;
  g1[0] = (int)g1w0;                                   // mask=0 | data_size | pad fields
  g1[1] = (int)(tdim0 << 16);                          // tensor_dim0[15:0]
  g1[2] = (int)((tdim0 >> 16) | (tdim1 << 16));        // tensor_dim0[31:16] | tensor_dim1[15:0]
  g1[3] = (int)((tdim1 >> 16) | (tile0 << 16));        // tensor_dim1[31:16] | tile_dim0
  g1[4] = (int)tile1;                                  // tile_dim1 | tile_dim2=0
  g1[5] = (int)stride0;                                // tensor_dim0_stride[31:0]
  g1[6] = 0;
  g1[7] = 0;
  i32x4 z4;
  z4[0] = z4[1] = z4[2] = z4[3] = 0;
  i32x8 z8;
#pragma unroll
  for (int i = 0; i < 8; ++i) z8[i] = 0;
  __builtin_amdgcn_tensor_load_to_lds(g0, g1, z4, z4, z8, 0);
}
// data_size=2B (code 1) | pad_enable | pad_interval | pad_amount(4 dwords -> code 3)
#define TDM_W0_K ((1u << 16) | (1u << 20) | (5u << 22) | (3u << 25))  // 64-DW rows (128 elt)
#define TDM_W0_V ((1u << 16) | (1u << 20) | (3u << 22) | (3u << 25))  // 16-DW rows (32 elt)
#endif

// ---------- fp32 -> bf16 convert ----------
__global__ __launch_bounds__(256) void cvt_kernel(const float* __restrict__ in,
                                                  __bf16* __restrict__ out, int n) {
  int i = blockIdx.x * 256 + threadIdx.x;
  if (i < n) out[i] = tobf(in[i]);
}

// ---------- fp32 [K][N] -> bf16 transposed [N][K] (tiled) ----------
__global__ __launch_bounds__(256) void cvtT_kernel(const float* __restrict__ in,
                                                   __bf16* __restrict__ out, int K, int N) {
  __shared__ float t[32][33];
  const int bx = blockIdx.x * 32;  // n
  const int by = blockIdx.y * 32;  // k
  const int lx = threadIdx.x & 31, ly = threadIdx.x >> 5;  // 8 rows per pass
#pragma unroll
  for (int i = 0; i < 32; i += 8)
    t[ly + i][lx] = in[(size_t)(by + ly + i) * N + bx + lx];
  __syncthreads();
#pragma unroll
  for (int i = 0; i < 32; i += 8)
    out[(size_t)(bx + ly + i) * K + by + lx] = tobf(t[lx][ly + i]);
}

// ---------- RoPE on q/k of fused qkv [S][6144]; v written transposed [g*128+d][S] ----------
__global__ __launch_bounds__(256) void rope_kernel(const __bf16* __restrict__ in,
                                                   __bf16* __restrict__ qk_out,
                                                   __bf16* __restrict__ vT_out) {
  int idx = blockIdx.x * 256 + threadIdx.x;       // 0 .. S*QKV_N-1
  int s = idx / QKV_N;
  int c = idx - s * QKV_N;
  float val = b2f(in[idx]);
  if (c < DMODEL + 1024) {                        // q (0..4095) or k (4096..5119)
    int hd = c & (HDIM - 1);
    int j = hd & 63;
    float invf = __expf(-(float)j * 0.16905903569391067f);   // 50000^(-j/64)
    float t = (float)s * invf;
    float cs = __cosf(t), sn = __sinf(t);
    float pv = b2f(in[(hd < 64) ? (idx + 64) : (idx - 64)]);
    float res = (hd < 64) ? (val * cs - pv * sn) : (val * cs + pv * sn);
    qk_out[idx] = tobf(res);
  } else {
    vT_out[(size_t)(c - (DMODEL + 1024)) * S_LEN + s] = tobf(val);
  }
}

// ---------- tiled bf16 WMMA GEMM: C[M,N] = A[M,K] @ Wt[N,K]^T + bias[N] ----------
// block 256 thr (8 waves as 2x4), tile 128(M) x 256(N), K-step 32, double-buffered
// async staging; wave tile 64x64 -> 16 wmma per wave per K-step.
template <bool F32OUT>
__global__ __launch_bounds__(256) void gemm_kernel(const __bf16* __restrict__ A,
                                                   const __bf16* __restrict__ Wt,
                                                   const float* __restrict__ bias,
                                                   void* __restrict__ Cout,
                                                   int K, int N) {
  __shared__ __bf16 At[2][128 * 40];   // [m][k], stride 40
  __shared__ __bf16 Bt[2][256 * 40];   // [n][k], stride 40 (Wt rows are contiguous in k)
  const int tid = threadIdx.x;
  const int wid = tid >> 5, lane = tid & 31;
  const int hi = lane >> 4, l15 = lane & 15;
  const int wm = wid >> 2, wn = wid & 3;
  const int m0 = blockIdx.y * 128, n0 = blockIdx.x * 256;
  const int sr = tid >> 2, skc = (tid & 3) * 8;   // staging: row, k-chunk

  v8f acc[4][4];
#pragma unroll
  for (int mt = 0; mt < 4; ++mt)
#pragma unroll
    for (int nt = 0; nt < 4; ++nt) acc[mt][nt] = vzero8();

  auto stage = [&](int buf, int k0) {
#pragma unroll
    for (int rep = 0; rep < 2; ++rep) {
      int r = sr + rep * 64;
      cp_async16(&At[buf][r * 40 + skc], A + (size_t)(m0 + r) * K + k0 + skc);
    }
#pragma unroll
    for (int rep = 0; rep < 4; ++rep) {
      int n = sr + rep * 64;
      cp_async16(&Bt[buf][n * 40 + skc], Wt + (size_t)(n0 + n) * K + k0 + skc);
    }
  };

  stage(0, 0);
  wait_async();
  __syncthreads();

  int buf = 0;
  for (int k0 = 0; k0 < K; k0 += 32) {
    if (k0 + 32 < K) stage(buf ^ 1, k0 + 32);   // overlap next-tile copy with WMMAs
    v16bf bfr[4];
#pragma unroll
    for (int nt = 0; nt < 4; ++nt) {
      const __bf16* p = &Bt[buf][(wn * 64 + nt * 16 + l15) * 40 + hi * 8];
      ((v8bf*)&bfr[nt])[0] = *(const v8bf*)p;
      ((v8bf*)&bfr[nt])[1] = *(const v8bf*)(p + 16);
    }
#pragma unroll
    for (int mt = 0; mt < 4; ++mt) {
      v16bf afr;
      const __bf16* p = &At[buf][(wm * 64 + mt * 16 + l15) * 40 + hi * 8];
      ((v8bf*)&afr)[0] = *(const v8bf*)p;
      ((v8bf*)&afr)[1] = *(const v8bf*)(p + 16);
#pragma unroll
      for (int nt = 0; nt < 4; ++nt)
        acc[mt][nt] = wmma_bf16(afr, bfr[nt], acc[mt][nt]);
    }
    wait_async();
    __syncthreads();
    buf ^= 1;
  }

#pragma unroll
  for (int mt = 0; mt < 4; ++mt)
#pragma unroll
    for (int nt = 0; nt < 4; ++nt) {
      int col = n0 + wn * 64 + nt * 16 + l15;
      float bc = bias[col];
#pragma unroll
      for (int j = 0; j < 8; ++j) {
        int row = m0 + wm * 64 + mt * 16 + j + 8 * hi;
        float v = acc[mt][nt][j] + bc;
        if constexpr (F32OUT)
          ((float*)Cout)[(size_t)row * N + col] = v;
        else
          ((__bf16*)Cout)[(size_t)row * N + col] = tobf(v);
      }
    }
}

// ---------- flash attention (causal, GQA), K/V tiles staged by the TDM ----------
// grid (NQH, S/128); block 256 = 8 waves; wave w owns query rows m0+16w .. +15
__global__ __launch_bounds__(256) void fa_kernel(const __bf16* __restrict__ qkv,
                                                 const __bf16* __restrict__ vT,
                                                 __bf16* __restrict__ attn_out) {
  __shared__ __bf16 Kt[32 * 136];        // [key][d], stride 136 (TDM pad: 64DW + 4DW)
  __shared__ __bf16 Vt[128 * 40];        // [d][key], stride 40  (TDM pad: 16DW + 4DW)
  __shared__ __bf16 Pt[8 * 16 * 40];     // per-wave P tile
  const int tid = threadIdx.x;
  const int wid = tid >> 5, lane = tid & 31;
  const int hi = lane >> 4, l15 = lane & 15;
  const int h = blockIdx.x, g = h >> 2;
  const int m0 = blockIdx.y * 128;
  const int r0 = m0 + wid * 16;

  const __bf16* qbase = qkv + h * HDIM;
  const __bf16* kbase = qkv + DMODEL + g * HDIM;
  const __bf16* vbase = vT + (size_t)(g * HDIM) * S_LEN;

  v16bf qf[4];
  {
    const __bf16* qrow = qbase + (size_t)(r0 + l15) * QKV_LD + hi * 8;
#pragma unroll
    for (int kc = 0; kc < 4; ++kc) {
      ((v8bf*)&qf[kc])[0] = *(const v8bf*)(qrow + kc * 32);
      ((v8bf*)&qf[kc])[1] = *(const v8bf*)(qrow + kc * 32 + 16);
    }
  }

  v8f o[8];
#pragma unroll
  for (int dt = 0; dt < 8; ++dt) o[dt] = vzero8();
  float mrow[8], srow[8];
#pragma unroll
  for (int j = 0; j < 8; ++j) { mrow[j] = -1e30f; srow[j] = 0.0f; }

  const float scale = 0.08838834764831845f;  // 1/sqrt(128)
  const int nchunks = (m0 + 128) >> 5;

  for (int ch = 0; ch < nchunks; ++ch) {
    const int kb = ch * 32;
    __syncthreads();  // all readers done with tiles before restage
#if HAVE_TDM
    if (wid == 0) {
      // K tile: 32 rows x 128 elts, row stride QKV_LD elements
      tdm_load_2d(kbase + (size_t)kb * QKV_LD, Kt, TDM_W0_K,
                  QKV_N, S_LEN, 128, 32, QKV_LD);
      // V^T tile: 128 rows x 32 elts, row stride S_LEN elements
      tdm_load_2d(vbase + kb, Vt, TDM_W0_V,
                  S_LEN, NKVH * HDIM, 32, 128, S_LEN);
      __builtin_amdgcn_s_wait_tensorcnt(0);
    }
#else
#pragma unroll
    for (int rep = 0; rep < 2; ++rep) {
      int c = tid + rep * 256;
      int key = c >> 4, dc = (c & 15) * 8;
      *(v8bf*)&Kt[key * 136 + dc] =
          *(const v8bf*)(kbase + (size_t)(kb + key) * QKV_LD + dc);
      int d = c >> 2, kc2 = (c & 3) * 8;
      *(v8bf*)&Vt[d * 40 + kc2] =
          *(const v8bf*)(vbase + (size_t)d * S_LEN + kb + kc2);
    }
#endif
    __syncthreads();

    v8f s0 = vzero8(), s1 = vzero8();
#pragma unroll
    for (int kc = 0; kc < 4; ++kc) {
      v16bf b0, b1;
      const __bf16* p0 = &Kt[l15 * 136 + kc * 32 + hi * 8];
      const __bf16* p1 = &Kt[(l15 + 16) * 136 + kc * 32 + hi * 8];
      ((v8bf*)&b0)[0] = *(const v8bf*)p0;  ((v8bf*)&b0)[1] = *(const v8bf*)(p0 + 16);
      ((v8bf*)&b1)[0] = *(const v8bf*)p1;  ((v8bf*)&b1)[1] = *(const v8bf*)(p1 + 16);
      s0 = wmma_bf16(qf[kc], b0, s0);
      s1 = wmma_bf16(qf[kc], b1, s1);
    }

#pragma unroll
    for (int j = 0; j < 8; ++j) {
      int row = r0 + j + 8 * hi;
      float a0 = s0[j] * scale; if (kb + l15 > row)      a0 = -1e30f;
      float a1 = s1[j] * scale; if (kb + 16 + l15 > row) a1 = -1e30f;
      float t = fmaxf(a0, a1);
      t = fmaxf(t, __shfl_xor(t, 1, 32));
      t = fmaxf(t, __shfl_xor(t, 2, 32));
      t = fmaxf(t, __shfl_xor(t, 4, 32));
      t = fmaxf(t, __shfl_xor(t, 8, 32));
      float mn = fmaxf(mrow[j], t);
      float alpha = __expf(mrow[j] - mn);
      mrow[j] = mn;
      float p0 = __expf(a0 - mn);
      float p1 = __expf(a1 - mn);
      Pt[wid * 640 + (j + 8 * hi) * 40 + l15]      = tobf(p0);
      Pt[wid * 640 + (j + 8 * hi) * 40 + 16 + l15] = tobf(p1);
      float ps = p0 + p1;
      ps += __shfl_xor(ps, 1, 32);
      ps += __shfl_xor(ps, 2, 32);
      ps += __shfl_xor(ps, 4, 32);
      ps += __shfl_xor(ps, 8, 32);
      srow[j] = srow[j] * alpha + ps;
#pragma unroll
      for (int dt = 0; dt < 8; ++dt) o[dt][j] *= alpha;
    }

    v16bf pf;
    {
      const __bf16* p = &Pt[wid * 640 + l15 * 40 + hi * 8];
      ((v8bf*)&pf)[0] = *(const v8bf*)p;
      ((v8bf*)&pf)[1] = *(const v8bf*)(p + 16);
    }
#pragma unroll
    for (int dt = 0; dt < 8; ++dt) {
      v16bf vf;
      const __bf16* p = &Vt[(dt * 16 + l15) * 40 + hi * 8];
      ((v8bf*)&vf)[0] = *(const v8bf*)p;
      ((v8bf*)&vf)[1] = *(const v8bf*)(p + 16);
      o[dt] = wmma_bf16(pf, vf, o[dt]);
    }
  }

  float inv[8];
#pragma unroll
  for (int j = 0; j < 8; ++j) inv[j] = 1.0f / srow[j];
#pragma unroll
  for (int dt = 0; dt < 8; ++dt)
#pragma unroll
    for (int j = 0; j < 8; ++j) {
      int row = r0 + j + 8 * hi;
      int col = h * HDIM + dt * 16 + l15;
      attn_out[(size_t)row * DMODEL + col] = tobf(o[dt][j] * inv[j]);
    }
}

extern "C" void kernel_launch(void* const* d_in, const int* in_sizes, int n_in,
                              void* d_out, int out_size, void* d_ws, size_t ws_size,
                              hipStream_t stream) {
  (void)in_sizes; (void)n_in; (void)out_size; (void)ws_size;
  const float* x    = (const float*)d_in[0];
  const float* Wqkv = (const float*)d_in[1];
  const float* bqkv = (const float*)d_in[2];
  const float* Wo   = (const float*)d_in[3];
  const float* bo   = (const float*)d_in[4];
  float* out = (float*)d_out;

  __bf16* xb     = (__bf16*)d_ws;                        // [S][D]       16 MB
  __bf16* wqkvT  = xb     + (size_t)S_LEN * DMODEL;      // [6144][4096] 48 MB
  __bf16* woT    = wqkvT  + (size_t)DMODEL * QKV_N;      // [4096][4096] 32 MB
  __bf16* qkvraw = woT    + (size_t)DMODEL * DMODEL;     // [S][6144]    24 MB
  __bf16* qkvr   = qkvraw + (size_t)S_LEN * QKV_N;       // q/k roped    24 MB
  __bf16* vT     = qkvr   + (size_t)S_LEN * QKV_N;       // [1024][S]     4 MB
  __bf16* attn   = vT     + (size_t)NKVH * HDIM * S_LEN; // [S][D]       16 MB

  { int n = S_LEN * DMODEL;  cvt_kernel<<<(n + 255) / 256, 256, 0, stream>>>(x, xb, n); }
  cvtT_kernel<<<dim3(QKV_N / 32, DMODEL / 32), 256, 0, stream>>>(Wqkv, wqkvT, DMODEL, QKV_N);
  cvtT_kernel<<<dim3(DMODEL / 32, DMODEL / 32), 256, 0, stream>>>(Wo, woT, DMODEL, DMODEL);

  gemm_kernel<false><<<dim3(QKV_N / 256, S_LEN / 128), 256, 0, stream>>>(
      xb, wqkvT, bqkv, (void*)qkvraw, DMODEL, QKV_N);

  { int n = S_LEN * QKV_N;
    rope_kernel<<<(n + 255) / 256, 256, 0, stream>>>(qkvraw, qkvr, vT); }

  fa_kernel<<<dim3(NQH, S_LEN / 128), 256, 0, stream>>>(qkvr, vT, attn);

  gemm_kernel<true><<<dim3(DMODEL / 256, S_LEN / 128), 256, 0, stream>>>(
      attn, woT, bo, (void*)out, DMODEL, DMODEL);
}
